// GPT_31636729103223
// MI455X (gfx1250) — compile-verified, matrix operand order checked by reading, not run on hardware
//
#include <hip/hip_runtime.h>
#include <hip/hip_bf16.h>

// ---------------------------------------------------------------------------
// GPT-small forward on gfx1250 (MI455X): bf16 WMMA GEMMs + flash attention,
// async global->LDS staging + TDM tensor_load_to_lds for attention K tiles.
// ---------------------------------------------------------------------------

typedef __bf16 bf16_t;
typedef bf16_t v16bf __attribute__((ext_vector_type(16)));
typedef bf16_t v8bf  __attribute__((ext_vector_type(8)));
typedef float  v8f   __attribute__((ext_vector_type(8)));
typedef unsigned short u16;
typedef u16          v8u16 __attribute__((ext_vector_type(8)));
typedef unsigned int v4u   __attribute__((ext_vector_type(4)));
typedef int          v4i   __attribute__((ext_vector_type(4)));
typedef int          v8i   __attribute__((ext_vector_type(8)));

#define TOK    2048   // BATCH*SEQ
#define SEQ    1024
#define BATCHN 2
#define DIM    768
#define HEADS  12
#define HD     64
#define FF     3072
#define VOCABN 50257
#define NBLK   6

#define MODE_BF16  0
#define MODE_HEADS 1
#define MODE_RESID 2
#define MODE_GELU  3
#define MODE_F32   4

__device__ __forceinline__ u16 f2bf(float f) {
  unsigned u = __float_as_uint(f);
  unsigned r = u + 0x7FFFu + ((u >> 16) & 1u);   // round-to-nearest-even
  return (u16)(r >> 16);
}
__device__ __forceinline__ v16bf cat16(v8bf lo, v8bf hi) {
  return __builtin_shufflevector(lo, hi, 0,1,2,3,4,5,6,7,8,9,10,11,12,13,14,15);
}
__device__ __forceinline__ v8f vzero8() {
  return (v8f){0.f,0.f,0.f,0.f,0.f,0.f,0.f,0.f};
}
// low 32 bits of a generic pointer into the LDS aperture == LDS byte offset
__device__ __forceinline__ unsigned lds_off(const void* p) {
  return (unsigned)(unsigned long long)p;
}
// 16B async copy global -> LDS (ASYNCcnt-tracked, bypasses VGPRs)
__device__ __forceinline__ void async_b128(unsigned lds_byte, const void* g) {
  unsigned long long ga = (unsigned long long)g;
  asm volatile("global_load_async_to_lds_b128 %0, %1, off"
               :: "v"(lds_byte), "v"(ga) : "memory");
}
__device__ __forceinline__ void wait_async() {
  asm volatile("s_wait_asynccnt 0x0" ::: "memory");
}

// ---------------------------------------------------------------------------
// Embedding: x[t,d] = tok_emb[tokens[t],d] + pos_emb[t%SEQ,d]   (f32)
// ---------------------------------------------------------------------------
__global__ __launch_bounds__(256) void k_embed(const int* __restrict__ tok,
                                               const float* __restrict__ te,
                                               const float* __restrict__ pe,
                                               float* __restrict__ x) {
  const int t  = blockIdx.x;
  const int s  = t & (SEQ - 1);
  const int id = tok[t];
  for (int d = threadIdx.x; d < DIM; d += 256)
    x[(size_t)t * DIM + d] = te[(size_t)id * DIM + d] + pe[(size_t)s * DIM + d];
}

// ---------------------------------------------------------------------------
// LayerNorm -> bf16 (GEMM A operand). One block per row of 768.
// ---------------------------------------------------------------------------
__global__ __launch_bounds__(256) void k_ln(const float* __restrict__ x,
                                            const float* __restrict__ g,
                                            const float* __restrict__ b,
                                            u16* __restrict__ y) {
  __shared__ float red[256];
  const int t = blockIdx.x;
  const float* row = x + (size_t)t * DIM;
  float v0[3], s = 0.f, sq = 0.f;
  #pragma unroll
  for (int i = 0; i < 3; ++i) {
    float v = row[threadIdx.x + 256 * i];
    v0[i] = v; s += v; sq += v * v;
  }
  red[threadIdx.x] = s; __syncthreads();
  for (int o = 128; o > 0; o >>= 1) {
    if (threadIdx.x < o) red[threadIdx.x] += red[threadIdx.x + o];
    __syncthreads();
  }
  const float mean = red[0] * (1.f / DIM);
  __syncthreads();
  red[threadIdx.x] = sq; __syncthreads();
  for (int o = 128; o > 0; o >>= 1) {
    if (threadIdx.x < o) red[threadIdx.x] += red[threadIdx.x + o];
    __syncthreads();
  }
  const float var = red[0] * (1.f / DIM) - mean * mean;
  const float rs  = rsqrtf(var + 1e-5f);
  #pragma unroll
  for (int i = 0; i < 3; ++i) {
    int d = threadIdx.x + 256 * i;
    y[(size_t)t * DIM + d] = f2bf((v0[i] - mean) * rs * g[d] + b[d]);
  }
}

// ---------------------------------------------------------------------------
// Tiled bf16 WMMA GEMM: C[M,N] = act(A_bf16[M,K] @ W_f32[K,N] + bias)
// block tile 64x128, 8 waves (4x2), wave tile 16x64, K-step 32.
// A tile staged by async global->LDS DMA; B tile f32->bf16 transposed [n][k]
// with pair-packed ds_store_b32.
// ---------------------------------------------------------------------------
__global__ __launch_bounds__(256) void k_gemm(const u16*  __restrict__ A,
                                              const float* __restrict__ W,
                                              const float* __restrict__ bias,
                                              int M, int N, int K, int mode,
                                              float* __restrict__ outf,
                                              u16*  __restrict__ outb) {
  __shared__ __align__(16) u16 sA[64 * 32];
  __shared__ __align__(16) u16 sB[128 * 32];
  const int tid  = threadIdx.x;
  const int lane = tid & 31, wave = tid >> 5;
  const int wr = wave >> 1, wc = wave & 1;
  const int m0 = blockIdx.y * 64, n0 = blockIdx.x * 128;
  const int l15 = lane & 15;
  const int kb  = (lane < 16) ? 0 : 8;

  v8f acc[4];
  #pragma unroll
  for (int t = 0; t < 4; ++t) acc[t] = vzero8();

  const int ar = tid >> 2, as4 = tid & 3;   // A stage: row, 8-elem segment
  const int kp = tid & 15, ns = tid >> 4;   // B stage: k-pair, 8-col segment
  const bool fastN = (n0 + 128 <= N) && ((N & 7) == 0);
  const bool fullN = (n0 + 128 <= N);

  for (int kk = 0; kk < K; kk += 32) {
    __syncthreads();
    // stage A (bf16): async DMA, 16B per thread
    async_b128(lds_off(sA + ar * 32 + as4 * 8),
               A + (size_t)(m0 + ar) * K + kk + as4 * 8);
    // stage B: rows 2kp,2kp+1; cols n0+ns*8..+7 ; pack bf16 pairs along k
    if (fastN) {
      const float* g0 = W + (size_t)(kk + 2 * kp) * N + n0 + ns * 8;
      v8f w0 = *(const v8f*)g0;
      v8f w1 = *(const v8f*)(g0 + N);
      #pragma unroll
      for (int j = 0; j < 8; ++j) {
        unsigned pk = (unsigned)f2bf(w0[j]) | ((unsigned)f2bf(w1[j]) << 16);
        *(unsigned*)(sB + (ns * 8 + j) * 32 + 2 * kp) = pk;
      }
    } else if (fullN) {                      // interior tiles, odd N (vocab)
      const float* g0 = W + (size_t)(kk + 2 * kp) * N + n0 + ns * 8;
      #pragma unroll
      for (int j = 0; j < 8; ++j) {
        unsigned pk = (unsigned)f2bf(g0[j]) | ((unsigned)f2bf(g0[N + j]) << 16);
        *(unsigned*)(sB + (ns * 8 + j) * 32 + 2 * kp) = pk;
      }
    } else {                                 // single tail tile
      #pragma unroll
      for (int j = 0; j < 8; ++j) {
        int n = n0 + ns * 8 + j;
        float a0 = 0.f, a1 = 0.f;
        if (n < N) {
          a0 = W[(size_t)(kk + 2 * kp) * N + n];
          a1 = W[(size_t)(kk + 2 * kp + 1) * N + n];
        }
        unsigned pkv = (unsigned)f2bf(a0) | ((unsigned)f2bf(a1) << 16);
        *(unsigned*)(sB + (ns * 8 + j) * 32 + 2 * kp) = pkv;
      }
    }
    wait_async();
    __syncthreads();

    const u16* pa = sA + (wr * 16 + l15) * 32 + kb;
    v16bf a = cat16(*(const v8bf*)pa, *(const v8bf*)(pa + 16));
    v16bf bfr[4];
    #pragma unroll
    for (int t = 0; t < 4; ++t) {            // batch ds_loads, then wmma chain
      const u16* pb = sB + (wc * 64 + t * 16 + l15) * 32 + kb;
      bfr[t] = cat16(*(const v8bf*)pb, *(const v8bf*)(pb + 16));
    }
    #pragma unroll
    for (int t = 0; t < 4; ++t)
      acc[t] = __builtin_amdgcn_wmma_f32_16x16x32_bf16(
          false, a, false, bfr[t], (short)0, acc[t], false, false);
  }

  // epilogue: C layout row = m0+wr*16 + e + 8*(lane>=16), col = sub + (lane&15)
  const int rhalf = (lane >> 4) << 3;
  #pragma unroll
  for (int t = 0; t < 4; ++t) {
    int n = n0 + wc * 64 + t * 16 + l15;
    if (n >= N) continue;
    float bsv = bias[n];
    #pragma unroll
    for (int e = 0; e < 8; ++e) {
      int m = m0 + wr * 16 + rhalf + e;
      float v = acc[t][e] + bsv;
      if (mode == MODE_BF16) {
        outb[(size_t)m * N + n] = f2bf(v);
      } else if (mode == MODE_HEADS) {          // scatter to [B,H,S,HD]
        int bI = m >> 10, sI = m & 1023, h = n >> 6, hd = n & 63;
        outb[(((size_t)(bI * HEADS + h)) * SEQ + sI) * HD + hd] = f2bf(v);
      } else if (mode == MODE_RESID) {          // x += proj
        outf[(size_t)m * N + n] += v;
      } else if (mode == MODE_GELU) {
        float gv = 0.5f * v * (1.f + erff(v * 0.70710678f));
        outb[(size_t)m * N + n] = f2bf(gv);
      } else {                                  // MODE_F32 (logits)
        outf[(size_t)m * N + n] = v;
      }
    }
  }
}

// ---------------------------------------------------------------------------
// Flash attention, causal. Q/K/V bf16 [B*H, S, 64]. Block = 128 query rows of
// one (b,h); wave = 16 rows. K tile fetched by the Tensor Data Mover
// (tensor_load_to_lds, 64x64x2B 2D tile descriptor), V tile transposed via
// pair-packed LDS stores. Q*K^T and P*V on v_wmma_f32_16x16x32_bf16.
// ---------------------------------------------------------------------------
__global__ __launch_bounds__(256) void k_attn(const u16* __restrict__ Q,
                                              const u16* __restrict__ Kb,
                                              const u16* __restrict__ Vb,
                                              u16* __restrict__ O) {
  __shared__ __align__(16) u16 sK [64 * 64];      // [key][hd]  (== B^T layout)
  __shared__ __align__(16) u16 sVt[64 * 64];      // [hd][key]
  __shared__ __align__(16) u16 sP [8 * 16 * 64];  // per-wave P staging
  const int tid  = threadIdx.x, lane = tid & 31, wave = tid >> 5;
  const int bh   = blockIdx.y;
  const int q0   = blockIdx.x * 128;
  const int bI   = bh / HEADS, h = bh % HEADS;
  const u16* Qp = Q  + (size_t)bh * SEQ * HD;
  const u16* Kp = Kb + (size_t)bh * SEQ * HD;
  const u16* Vp = Vb + (size_t)bh * SEQ * HD;
  const int l15 = lane & 15;
  const int kbo = (lane < 16) ? 0 : 8;
  const int rhalf = (lane >> 4) << 3;

  // preload Q fragments (Hd=64 -> 2 K-steps of 32)
  const int qrow = q0 + wave * 16 + l15;
  v16bf aq[2];
  #pragma unroll
  for (int ks = 0; ks < 2; ++ks) {
    const u16* p = Qp + (size_t)qrow * HD + ks * 32 + kbo;
    aq[ks] = cat16(*(const v8bf*)p, *(const v8bf*)(p + 16));
  }

  float mrow[8], lrow[8];
  v8f o[4];
  #pragma unroll
  for (int e = 0; e < 8; ++e) { mrow[e] = -1e30f; lrow[e] = 0.f; }
  #pragma unroll
  for (int t = 0; t < 4; ++t) o[t] = vzero8();

  const int kp2 = tid & 31;   // V stage: key pair 2kp2,2kp2+1
  const int hs  = tid >> 5;   // V stage: hd segment of 8

  for (int j0 = 0; j0 < q0 + 128; j0 += 64) {
    __syncthreads();
    // ---- K tile via Tensor Data Mover: 2D tile 64(hd) x 64(key), 2B elems
    if (wave == 0) {
      unsigned long long ga = (unsigned long long)(const void*)(Kp + (size_t)j0 * HD);
      v4u g0;
      g0[0] = 1u;                                   // count=1, user mode
      g0[1] = lds_off(sK);                          // LDS byte address
      g0[2] = (unsigned)(ga & 0xffffffffu);         // global_addr[31:0]
      g0[3] = (unsigned)((ga >> 32) & 0x01ffffffu)  // global_addr[56:32]
              | (2u << 30);                         // type=2 (image)
      v8i g1 = { 0x00020000,        // data_size=2B (code 2<<16, bits17:16=2)
                 (64 << 16),        // tensor_dim0[15:0]=64 @ bits63:48
                 (SEQ << 16),       // tensor_dim1[15:0]=1024 @ bits95:80
                 (64 << 16),        // tile_dim0=64 @ bits127:112
                 64,                // tile_dim1=64, tile_dim2=0
                 64,                // tensor_dim0_stride[31:0]=64
                 0, 0 };
      v4i gz = { 0, 0, 0, 0 };
#if defined(__clang_major__) && (__clang_major__ >= 23)
      v8i gz8 = { 0, 0, 0, 0, 0, 0, 0, 0 };
      __builtin_amdgcn_tensor_load_to_lds(g0, g1, gz, gz, gz8, 0);
#else
      __builtin_amdgcn_tensor_load_to_lds(g0, g1, gz, gz, 0);
#endif
    }
    // ---- V tile transposed: two 16B row loads, pack key-pairs, b32 stores
    {
      const u16* r0 = Vp + (size_t)(j0 + 2 * kp2) * HD + hs * 8;
      v8u16 e0 = *(const v8u16*)r0;
      v8u16 e1 = *(const v8u16*)(r0 + HD);
      #pragma unroll
      for (int j = 0; j < 8; ++j) {
        unsigned pk = (unsigned)e0[j] | ((unsigned)e1[j] << 16);
        *(unsigned*)(sVt + (hs * 8 + j) * 64 + 2 * kp2) = pk;
      }
    }
    if (wave == 0) __builtin_amdgcn_s_wait_tensorcnt(0);
    __syncthreads();

    // scores S = Q K^T
    v8f sc[4];
    #pragma unroll
    for (int t = 0; t < 4; ++t) sc[t] = vzero8();
    #pragma unroll
    for (int ks = 0; ks < 2; ++ks) {
      v16bf bfr[4];
      #pragma unroll
      for (int t = 0; t < 4; ++t) {
        const u16* pb = sK + (t * 16 + l15) * 64 + ks * 32 + kbo;
        bfr[t] = cat16(*(const v8bf*)pb, *(const v8bf*)(pb + 16));
      }
      #pragma unroll
      for (int t = 0; t < 4; ++t)
        sc[t] = __builtin_amdgcn_wmma_f32_16x16x32_bf16(
            false, aq[ks], false, bfr[t], (short)0, sc[t], false, false);
    }

    // scale + causal mask + running max
    float mnew[8];
    #pragma unroll
    for (int e = 0; e < 8; ++e) mnew[e] = mrow[e];
    #pragma unroll
    for (int t = 0; t < 4; ++t) {
      int col = j0 + t * 16 + l15;
      #pragma unroll
      for (int e = 0; e < 8; ++e) {
        int rowm = q0 + wave * 16 + rhalf + e;
        float v = sc[t][e] * 0.125f;                    // 1/sqrt(64)
        if (col > rowm) v = -1e30f;
        sc[t][e] = v;
        mnew[e] = fmaxf(mnew[e], v);
      }
    }
    #pragma unroll
    for (int e = 0; e < 8; ++e) {                       // reduce across 16 cols
      float m = mnew[e];
      m = fmaxf(m, __shfl_xor(m, 1, 32));
      m = fmaxf(m, __shfl_xor(m, 2, 32));
      m = fmaxf(m, __shfl_xor(m, 4, 32));
      m = fmaxf(m, __shfl_xor(m, 8, 32));
      mnew[e] = m;
    }
    float resc[8], psum[8];
    #pragma unroll
    for (int e = 0; e < 8; ++e) {
      resc[e] = __expf(mrow[e] - mnew[e]);
      mrow[e] = mnew[e];
      psum[e] = 0.f;
    }
    #pragma unroll
    for (int t = 0; t < 4; ++t)
      #pragma unroll
      for (int e = 0; e < 8; ++e) {
        float p = __expf(sc[t][e] - mrow[e]);
        sc[t][e] = p;
        psum[e] += p;
      }
    #pragma unroll
    for (int e = 0; e < 8; ++e) {
      float s = psum[e];
      s += __shfl_xor(s, 1, 32); s += __shfl_xor(s, 2, 32);
      s += __shfl_xor(s, 4, 32); s += __shfl_xor(s, 8, 32);
      lrow[e] = lrow[e] * resc[e] + s;
    }
    #pragma unroll
    for (int t = 0; t < 4; ++t)
      #pragma unroll
      for (int e = 0; e < 8; ++e) o[t][e] *= resc[e];

    // P: C-layout -> LDS -> A-layout, then O += P V
    u16* Pw = sP + wave * 16 * 64;
    #pragma unroll
    for (int t = 0; t < 4; ++t)
      #pragma unroll
      for (int e = 0; e < 8; ++e)
        Pw[(rhalf + e) * 64 + t * 16 + l15] = f2bf(sc[t][e]);
    #pragma unroll
    for (int ks = 0; ks < 2; ++ks) {
      const u16* pa = Pw + l15 * 64 + ks * 32 + kbo;
      v16bf ap = cat16(*(const v8bf*)pa, *(const v8bf*)(pa + 16));
      v16bf bfr[4];
      #pragma unroll
      for (int t = 0; t < 4; ++t) {
        const u16* pb = sVt + (t * 16 + l15) * 64 + ks * 32 + kbo;
        bfr[t] = cat16(*(const v8bf*)pb, *(const v8bf*)(pb + 16));
      }
      #pragma unroll
      for (int t = 0; t < 4; ++t)
        o[t] = __builtin_amdgcn_wmma_f32_16x16x32_bf16(
            false, ap, false, bfr[t], (short)0, o[t], false, false);
    }
  }

  // write normalized output gathered back to [T, DIM]
  #pragma unroll
  for (int e = 0; e < 8; ++e) {
    int rowm = q0 + wave * 16 + rhalf + e;
    float inv = 1.f / lrow[e];
    size_t base = ((size_t)(bI * SEQ + rowm)) * DIM + h * HD;
    #pragma unroll
    for (int t = 0; t < 4; ++t)
      O[base + t * 16 + l15] = f2bf(o[t][e] * inv);
  }
}

// ---------------------------------------------------------------------------
// Host orchestration. Blocks dict leaves assumed in JAX sorted-key order:
// W1,W2,Wk,Wp,Wq,Wv,b1,b2,bk,bp,bq,bv,ln1_b,ln1_s,ln2_b,ln2_s  (stacked [6,...])
// ---------------------------------------------------------------------------
extern "C" void kernel_launch(void* const* d_in, const int* in_sizes, int n_in,
                              void* d_out, int out_size, void* d_ws, size_t ws_size,
                              hipStream_t stream) {
  const int*   tokens  = (const int*)d_in[0];
  const float* tok_emb = (const float*)d_in[1];
  const float* pos_emb = (const float*)d_in[2];
  const float* W1  = (const float*)d_in[3];
  const float* W2  = (const float*)d_in[4];
  const float* Wk  = (const float*)d_in[5];
  const float* Wp  = (const float*)d_in[6];
  const float* Wq  = (const float*)d_in[7];
  const float* Wv  = (const float*)d_in[8];
  const float* b1  = (const float*)d_in[9];
  const float* b2  = (const float*)d_in[10];
  const float* bk  = (const float*)d_in[11];
  const float* bp  = (const float*)d_in[12];
  const float* bq  = (const float*)d_in[13];
  const float* bv  = (const float*)d_in[14];
  const float* ln1_b = (const float*)d_in[15];
  const float* ln1_s = (const float*)d_in[16];
  const float* ln2_b = (const float*)d_in[17];
  const float* ln2_s = (const float*)d_in[18];
  const float* lnf_s = (const float*)d_in[19];
  const float* lnf_b = (const float*)d_in[20];
  const float* Wout  = (const float*)d_in[21];
  const float* bout  = (const float*)d_in[22];

  char* wsb = (char*)d_ws;
  size_t off = 0;
  auto carve = [&](size_t bytes) -> void* {
    void* p = wsb + off;
    off += (bytes + 255) & ~(size_t)255;
    return p;
  };
  float* x   = (float*)carve((size_t)TOK * DIM * 4);  // residual stream f32
  u16* hbf   = (u16*)carve((size_t)TOK * DIM * 2);    // LN output bf16
  u16* qbuf  = (u16*)carve((size_t)TOK * DIM * 2);    // [B,H,S,64]
  u16* kbuf  = (u16*)carve((size_t)TOK * DIM * 2);
  u16* vbuf  = (u16*)carve((size_t)TOK * DIM * 2);
  u16* obuf  = (u16*)carve((size_t)TOK * DIM * 2);    // attn out [T,DIM]
  u16* mbuf  = (u16*)carve((size_t)TOK * FF * 2);     // MLP mid bf16

  const dim3 blk(256);
  const dim3 gD((DIM + 127) / 128, TOK / 64);         // (6, 32)
  const dim3 gF((FF + 127) / 128, TOK / 64);          // (24, 32)
  const dim3 gV((VOCABN + 127) / 128, TOK / 64);      // (393, 32)
  const dim3 gA(SEQ / 128, BATCHN * HEADS);           // (8, 24)

  k_embed<<<TOK, blk, 0, stream>>>(tokens, tok_emb, pos_emb, x);

  for (int i = 0; i < NBLK; ++i) {
    const size_t wdd = (size_t)i * DIM * DIM;
    k_ln<<<TOK, blk, 0, stream>>>(x, ln1_s + i * DIM, ln1_b + i * DIM, hbf);
    k_gemm<<<gD, blk, 0, stream>>>(hbf, Wq + wdd, bq + i * DIM,
                                   TOK, DIM, DIM, MODE_HEADS, nullptr, qbuf);
    k_gemm<<<gD, blk, 0, stream>>>(hbf, Wk + wdd, bk + i * DIM,
                                   TOK, DIM, DIM, MODE_HEADS, nullptr, kbuf);
    k_gemm<<<gD, blk, 0, stream>>>(hbf, Wv + wdd, bv + i * DIM,
                                   TOK, DIM, DIM, MODE_HEADS, nullptr, vbuf);
    k_attn<<<gA, blk, 0, stream>>>(qbuf, kbuf, vbuf, obuf);
    k_gemm<<<gD, blk, 0, stream>>>(obuf, Wp + wdd, bp + i * DIM,
                                   TOK, DIM, DIM, MODE_RESID, x, nullptr);
    k_ln<<<TOK, blk, 0, stream>>>(x, ln2_s + i * DIM, ln2_b + i * DIM, hbf);
    k_gemm<<<gF, blk, 0, stream>>>(hbf, W1 + (size_t)i * DIM * FF, b1 + i * FF,
                                   TOK, FF, DIM, MODE_GELU, nullptr, mbuf);
    k_gemm<<<gD, blk, 0, stream>>>(mbuf, W2 + (size_t)i * FF * DIM, b2 + i * DIM,
                                   TOK, DIM, FF, MODE_RESID, x, nullptr);
  }

  k_ln<<<TOK, blk, 0, stream>>>(x, lnf_s, lnf_b, hbf);
  k_gemm<<<gV, blk, 0, stream>>>(hbf, Wout, bout,
                                 TOK, VOCABN, DIM, MODE_F32, (float*)d_out, nullptr);
}
